// DIVeR_60610578481275
// MI455X (gfx1250) — matrix operand order repaired and checked by reading, not compiled, not run on hardware
//
#include <hip/hip_runtime.h>
#include <hip/hip_bf16.h>

typedef __attribute__((ext_vector_type(16))) _Float16 v16h;
typedef __attribute__((ext_vector_type(8)))  _Float16 v8h;
typedef __attribute__((ext_vector_type(8)))  float    v8f;

#define VOXEL_NUM   128
#define VOXEL_DIM   32
#define VOXEL_SIZE  0.015625f            // 2.0 / 128
#define XYZMAX_C    1.0f
#define XYZMIN_C   (-1.0f)
#define BRAYS       1024
#define KPTS        389                  // 3*(128+1)+2
#define MSEG        388
#define BIGV        1.0e10f
#define SX          264                  // LDS row stride in halves (16B aligned)
#define TILE_ROWS   256                  // points per block (8 waves x 32 points)

// ---------------------------------------------------------------------------
// Weight packing: f32 -> f16, transposed to [N][K] with zero padding.
// wt layout (halves): wt0 [256][32]@0, wt1 [256][256]@8192, wt2 [80][256]@73728,
//                     wt3 [256][96]@94208, wt4 [256][256]@118784, wt5 [16][256]@184320
// wt3 K-mapping: kk=0,1 -> 0 (cancels parked sigma/beta logits in X2 cols 0/1),
//                kk=2..92 -> w3 row kk-2 (h in cols 2..65, posenc in cols 66..92),
//                kk=93..95 -> 0.
// bias layout (f32):  b0@0(256) b1@256(256) b2@512(80) b3@592(256) b4@848(256) b5@1104(16)
// ---------------------------------------------------------------------------
__global__ void pack_weights(const float* __restrict__ w0, const float* __restrict__ w1,
                             const float* __restrict__ w2, const float* __restrict__ w3,
                             const float* __restrict__ w4, const float* __restrict__ w5,
                             const float* __restrict__ b0, const float* __restrict__ b1,
                             const float* __restrict__ b2, const float* __restrict__ b3,
                             const float* __restrict__ b4, const float* __restrict__ b5,
                             _Float16* __restrict__ wt, float* __restrict__ bias)
{
    int t = blockIdx.x * blockDim.x + threadIdx.x;
    if (t < 8192)  { int n = t / 32,  k = t % 32;  wt[t]          = (_Float16)w0[k * 256 + n]; }
    if (t < 65536) { int n = t / 256, k = t % 256; wt[8192 + t]   = (_Float16)w1[k * 256 + n]; }
    if (t < 20480) { int n = t / 256, k = t % 256; wt[73728 + t]  = (_Float16)((n < 66) ? w2[k * 66 + n] : 0.f); }
    if (t < 24576) { int n = t / 96,  kk = t % 96;
                     wt[94208 + t]  = (_Float16)((kk >= 2 && kk < 93) ? w3[(kk - 2) * 256 + n] : 0.f); }
    if (t < 65536) { int n = t / 256, k = t % 256; wt[118784 + t] = (_Float16)w4[k * 256 + n]; }
    if (t < 4096)  { int n = t / 256, k = t % 256; wt[184320 + t] = (_Float16)((n < 3) ? w5[k * 3 + n] : 0.f); }
    if (t < 256) { bias[t] = b0[t]; bias[256 + t] = b1[t]; bias[592 + t] = b3[t]; bias[848 + t] = b4[t]; }
    if (t < 80)  bias[512 + t]  = (t < 66) ? b2[t] : 0.f;
    if (t < 16)  bias[1104 + t] = (t < 3)  ? b5[t] : 0.f;
}

// ---------------------------------------------------------------------------
// Ray/voxel intersection + per-ray bitonic sort of 389 t-values (padded 512).
// ---------------------------------------------------------------------------
__global__ void intersect_kernel(const float* __restrict__ os, const float* __restrict__ ds_,
                                 float* __restrict__ ws_ts, float* __restrict__ ws_coord,
                                 float* __restrict__ ws_seg, float* __restrict__ ws_venc,
                                 float* __restrict__ out, int P)
{
    __shared__ float ts[512];
    const int b = blockIdx.x;
    const int tid = threadIdx.x;

    float oj[3], dj[3], inv[3];
    float tnear = -3.4e38f, tfar = 3.4e38f;
    #pragma unroll
    for (int j = 0; j < 3; ++j) {
        oj[j] = os[b * 3 + j];
        dj[j] = ds_[b * 3 + j];
        float dsafe = (fabsf(dj[j]) < 1e-9f) ? 1e-9f : dj[j];
        inv[j] = 1.0f / dsafe;
        float t0 = (XYZMIN_C - oj[j]) * inv[j];
        float t1 = (XYZMAX_C - oj[j]) * inv[j];
        tnear = fmaxf(tnear, fminf(t0, t1));
        tfar  = fminf(tfar,  fmaxf(t0, t1));
    }
    float tmin = fmaxf(tnear, 0.0f);
    bool  hit  = tfar > tmin;

    for (int i = tid; i < 512; i += blockDim.x) {
        float v;
        if (i == 0)        v = tmin;
        else if (i == 1)   v = tfar;
        else if (i < KPTS) { int t = i - 2; int j = t / 129; int pi = t % 129;
                             v = (XYZMIN_C + VOXEL_SIZE * (float)pi - oj[j]) * inv[j]; }
        else               v = BIGV;
        if (i < KPTS) {
            bool valid = (v >= tmin - 1e-6f) && (v <= tfar + 1e-6f) && hit;
            if (!valid) v = BIGV;
        }
        ts[i] = v;
    }
    __syncthreads();

    for (int k = 2; k <= 512; k <<= 1) {
        for (int j = k >> 1; j > 0; j >>= 1) {
            for (int i = tid; i < 512; i += blockDim.x) {
                int ixj = i ^ j;
                if (ixj > i) {
                    bool up = ((i & k) == 0);
                    float a = ts[i], c = ts[ixj];
                    if ((a > c) == up) { ts[i] = c; ts[ixj] = a; }
                }
            }
            __syncthreads();
        }
    }

    for (int i = tid; i < KPTS; i += blockDim.x) ws_ts[b * KPTS + i] = ts[i];
    for (int t = tid; t < KPTS * 3; t += blockDim.x) {
        int i = t / 3, j = t % 3;
        float pt = oj[j] + ts[i] * dj[j];
        float c = (pt - XYZMIN_C) * (1.0f / VOXEL_SIZE);
        c = fminf(fmaxf(c, 0.0f), 128.0f);
        ws_coord[(b * KPTS + i) * 3 + j] = c;
    }
    for (int m = tid; m < MSEG; m += blockDim.x) {
        float t0 = ts[m], t1 = ts[m + 1];
        bool pm0 = t0 < 0.5f * BIGV, pm1 = t1 < 0.5f * BIGV;
        float sm = (pm0 && pm1 && (t1 - t0 > 1e-6f)) ? 1.0f : 0.0f;
        ws_seg[b * MSEG + m]       = sm;
        out[5 * P + b * MSEG + m]  = sm;               // segmask
        out[6 * P + b * MSEG + m]  = pm0 ? t0 : 0.0f;  // ts_out
    }
    if (tid < 27) {   // posenc: [d(3), sin(12 freq-major), cos(12)]
        float v;
        if (tid < 3) v = dj[tid];
        else if (tid < 15) { int t = tid - 3;  int f = t / 3, j = t % 3; v = sinf(dj[j] * (float)(1 << f)); }
        else               { int t = tid - 15; int f = t / 3, j = t % 3; v = cosf(dj[j] * (float)(1 << f)); }
        ws_venc[b * 32 + tid] = v;
    }
}

// ---------------------------------------------------------------------------
// Simpson-weighted trilinear gather of 32-dim voxel features -> f16 rows.
// ---------------------------------------------------------------------------
__global__ void integrate_kernel(const float* __restrict__ coord, const float* __restrict__ seg,
                                 const float* __restrict__ voxels, _Float16* __restrict__ feat, int P)
{
    int p = blockIdx.x * blockDim.x + threadIdx.x;
    if (p >= P) return;
    int b = p / MSEG, m = p % MSEG;
    const float* cinp  = coord + (size_t)(b * KPTS + m) * 3;
    const float* coutp = cinp + 3;

    float a[3], bb[3], mm[3];
    int idx[3];
    float len2 = 0.0f;
    #pragma unroll
    for (int j = 0; j < 3; ++j) {
        float ci = cinp[j], co = coutp[j];
        int id = (int)floorf(0.5f * (ci + co));
        id = min(max(id, 0), VOXEL_NUM - 1);
        idx[j] = id;
        a[j]  = ci - (float)id;
        bb[j] = co - (float)id;
        mm[j] = 0.5f * (a[j] + bb[j]);
        float dd = co - ci;
        len2 += dd * dd;
    }
    float length = sqrtf(len2) * VOXEL_SIZE;
    float sm = seg[p];

    float facc[VOXEL_DIM];
    #pragma unroll
    for (int ch = 0; ch < VOXEL_DIM; ++ch) facc[ch] = 0.0f;

    for (int c = 0; c < 8; ++c) {
        int cc[3] = { (c >> 2) & 1, (c >> 1) & 1, c & 1 };
        float wA = 1.f, wM = 1.f, wB = 1.f;
        #pragma unroll
        for (int j = 0; j < 3; ++j) {
            float cj = (float)cc[j];
            wA *= cj * a[j]  + (1.f - cj) * (1.f - a[j]);
            wM *= cj * mm[j] + (1.f - cj) * (1.f - mm[j]);
            wB *= cj * bb[j] + (1.f - cj) * (1.f - bb[j]);
        }
        float wgt = (wA + 4.f * wM + wB) * (1.f / 6.f) * length * sm;
        const float* vox = voxels +
            ((((size_t)(idx[0] + cc[0])) * 129 + (size_t)(idx[1] + cc[1])) * 129 + (size_t)(idx[2] + cc[2])) * VOXEL_DIM;
        #pragma unroll
        for (int ch = 0; ch < VOXEL_DIM; ++ch) facc[ch] += wgt * vox[ch];
    }
    #pragma unroll
    for (int ch = 0; ch < VOXEL_DIM; ++ch) feat[(size_t)p * VOXEL_DIM + ch] = (_Float16)facc[ch];
}

// ---------------------------------------------------------------------------
// WMMA fused MLP. Weights are the A operand (M = neuron), activations the B
// operand (N = lane = point): each lane's 8 accumulator values are 8
// consecutive neurons of one point's row -> single 16B LDS store per tile.
// ---------------------------------------------------------------------------
__device__ __forceinline__ v16h load_w_frag(const _Float16* __restrict__ Wt, int Kdim,
                                            int nt, int kb, int hl, int hi)
{
    // A-matrix 16x32 layout: lane row M = nt+hl; K = kb + hi*16 + e (16 contiguous halves).
    const _Float16* p = Wt + (size_t)(nt + hl) * Kdim + kb + hi * 16;
    v8h lo = *(const v8h*)p;
    v8h hv = *(const v8h*)(p + 8);
    v16h a;
    #pragma unroll
    for (int e = 0; e < 8; ++e) { a[e] = lo[e]; a[e + 8] = hv[e]; }
    return a;
}

__device__ __forceinline__ v16h load_x_frag(const _Float16* __restrict__ X, int point, int kb, int hi)
{
    // B-matrix 32x16 layout: lane col N = point; K = kb + hi*16 + e (16 contiguous halves).
    const _Float16* p = X + point * SX + kb + hi * 16;
    v8h b0 = *(const v8h*)p;
    v8h b1 = *(const v8h*)(p + 8);
    v16h b;
    #pragma unroll
    for (int e = 0; e < 8; ++e) { b[e] = b0[e]; b[e + 8] = b1[e]; }
    return b;
}

__device__ __forceinline__ void gemm_layer(const _Float16* __restrict__ X, _Float16* __restrict__ Y,
                                           const _Float16* __restrict__ Wt, const float* __restrict__ bias,
                                           int Kdim, int Nout, bool relu, int lane, int wid)
{
    const int hl = lane & 15, hi = lane >> 4;
    const int p0 = wid * 32 + hl;       // first point-group
    const int p1 = p0 + 16;             // second point-group
    for (int nt = 0; nt < Nout; nt += 16) {
        v8f acc0 = {}, acc1 = {};
        for (int kb = 0; kb < Kdim; kb += 32) {
            v16h wf = load_w_frag(Wt, Kdim, nt, kb, hl, hi);
            v16h x0 = load_x_frag(X, p0, kb, hi);
            v16h x1 = load_x_frag(X, p1, kb, hi);
            acc0 = __builtin_amdgcn_wmma_f32_16x16x32_f16(false, wf, false, x0, (short)0, acc0, false, false);
            acc1 = __builtin_amdgcn_wmma_f32_16x16x32_f16(false, wf, false, x1, (short)0, acc1, false, false);
        }
        const int nb = nt + hi * 8;     // C/D layout: vgpr i -> neuron nb+i, lane -> point
        v8h o0 = {}, o1 = {};
        #pragma unroll
        for (int i = 0; i < 8; ++i) {
            float bv = bias[nb + i];
            float v0 = acc0[i] + bv;
            float v1 = acc1[i] + bv;
            if (relu) { v0 = fmaxf(v0, 0.0f); v1 = fmaxf(v1, 0.0f); }
            o0[i] = (_Float16)v0;
            o1[i] = (_Float16)v1;
        }
        *(v8h*)(Y + p0 * SX + nb) = o0;   // one 16B LDS store per tile
        *(v8h*)(Y + p1 * SX + nb) = o1;
    }
}

__device__ __forceinline__ float softplus_f(float x) { return (x > 20.f) ? x : log1pf(expf(x)); }
__device__ __forceinline__ float sigmoid_f(float x)  { return 1.0f / (1.0f + expf(-x)); }

__global__ void fused_mlp(const _Float16* __restrict__ feat, const _Float16* __restrict__ wt,
                          const float* __restrict__ bias, const float* __restrict__ seg,
                          const float* __restrict__ venc, float* __restrict__ out, int P)
{
    extern __shared__ char smem[];
    _Float16* Xa = (_Float16*)smem;                         // TILE_ROWS x SX halves
    _Float16* Xb = (_Float16*)(smem + TILE_ROWS * SX * 2);  // TILE_ROWS x SX halves

    const int tid = threadIdx.x;
    const int lane = tid & 31, wid = tid >> 5;
    const int tileBase = blockIdx.x * TILE_ROWS;

    // warm L2/WGP$ with the packed weights (gfx1250 global_prefetch path)
    for (size_t off = (size_t)tid * 256; off < 376832; off += (size_t)blockDim.x * 256)
        __builtin_prefetch((const char*)wt + off, 0, 1);

    // load feature tile (K=32), zero-fill out-of-range rows
    for (int i = tid; i < TILE_ROWS * 32; i += blockDim.x) {
        int r = i >> 5, k = i & 31;
        int p = tileBase + r;
        Xa[r * SX + k] = (p < P) ? feat[(size_t)p * 32 + k] : (_Float16)0.0f;
    }
    __syncthreads();

    // MLP1 — no barriers between layers: each wave only touches its own 32 rows.
    gemm_layer(Xa, Xb, wt + 0,      bias + 0,    32,  256, true,  lane, wid);
    gemm_layer(Xb, Xa, wt + 8192,   bias + 256,  256, 256, true,  lane, wid);
    gemm_layer(Xa, Xb, wt + 73728,  bias + 512,  256, 80,  false, lane, wid);
    __syncthreads();

    // sigma/beta from raw logits in cols 0/1 (zero-weighted in Wt3), posenc into cols 66..95.
    for (int r = tid; r < TILE_ROWS; r += blockDim.x) {
        int p = tileBase + r;
        if (p < P) {
            float fm = seg[p];
            out[3 * P + p] = softplus_f((float)Xb[r * SX + 0]) * fm;   // sigma
            out[4 * P + p] = softplus_f((float)Xb[r * SX + 1]) * fm;   // beta
        }
    }
    for (int i = tid; i < TILE_ROWS * 32; i += blockDim.x) {
        int r = i >> 5, c = 64 + (i & 31);
        if (c >= 66) {                       // cols 64,65 still hold h (k=62,63)
            int p = tileBase + r;
            _Float16 v = (_Float16)0.0f;
            if (p < P && c < 93) v = (_Float16)venc[(p / MSEG) * 32 + (c - 66)];
            Xb[r * SX + c] = v;
        }
    }
    __syncthreads();

    // MLP2
    gemm_layer(Xb, Xa, wt + 94208,  bias + 592,  96,  256, true,  lane, wid);
    gemm_layer(Xa, Xb, wt + 118784, bias + 848,  256, 256, true,  lane, wid);
    gemm_layer(Xb, Xa, wt + 184320, bias + 1104, 256, 16,  false, lane, wid);
    __syncthreads();

    for (int i = tid; i < TILE_ROWS * 3; i += blockDim.x) {
        int r = i / 3, c = i % 3;
        int p = tileBase + r;
        if (p < P) {
            float fm = seg[p];
            out[p * 3 + c] = sigmoid_f((float)Xa[r * SX + c]) * fm;    // color
        }
    }
}

// ---------------------------------------------------------------------------
extern "C" void kernel_launch(void* const* d_in, const int* in_sizes, int n_in,
                              void* d_out, int out_size, void* d_ws, size_t ws_size,
                              hipStream_t stream)
{
    const float* os     = (const float*)d_in[0];
    const float* ds_    = (const float*)d_in[1];
    const float* voxels = (const float*)d_in[2];
    const float* m1w0   = (const float*)d_in[3];
    const float* m1b0   = (const float*)d_in[4];
    const float* m1w1   = (const float*)d_in[5];
    const float* m1b1   = (const float*)d_in[6];
    const float* m1w2   = (const float*)d_in[7];
    const float* m1b2   = (const float*)d_in[8];
    const float* m2w0   = (const float*)d_in[9];
    const float* m2b0   = (const float*)d_in[10];
    const float* m2w1   = (const float*)d_in[11];
    const float* m2b1   = (const float*)d_in[12];
    const float* m2w2   = (const float*)d_in[13];
    const float* m2b2   = (const float*)d_in[14];

    const int P = BRAYS * MSEG;

    char* ws = (char*)d_ws;
    size_t off = 0;
    auto take = [&](size_t bytes) { size_t o = off; off += (bytes + 255) & ~(size_t)255; return o; };
    float*    ws_ts    = (float*)(ws + take((size_t)BRAYS * KPTS * 4));
    float*    ws_coord = (float*)(ws + take((size_t)BRAYS * KPTS * 3 * 4));
    float*    ws_seg   = (float*)(ws + take((size_t)P * 4));
    float*    ws_venc  = (float*)(ws + take((size_t)BRAYS * 32 * 4));
    _Float16* ws_feat  = (_Float16*)(ws + take((size_t)P * 32 * 2));
    _Float16* ws_wt    = (_Float16*)(ws + take((size_t)188416 * 2));
    float*    ws_bias  = (float*)(ws + take((size_t)1120 * 4));

    pack_weights<<<256, 256, 0, stream>>>(m1w0, m1w1, m1w2, m2w0, m2w1, m2w2,
                                          m1b0, m1b1, m1b2, m2b0, m2b1, m2b2,
                                          ws_wt, ws_bias);
    intersect_kernel<<<BRAYS, 256, 0, stream>>>(os, ds_, ws_ts, ws_coord, ws_seg, ws_venc,
                                                (float*)d_out, P);
    integrate_kernel<<<(P + 255) / 256, 256, 0, stream>>>(ws_coord, ws_seg, voxels, ws_feat, P);

    const size_t smemBytes = (size_t)2 * TILE_ROWS * SX * 2;   // Xa + Xb = 270336 B (<= 320KB WGP LDS)
    fused_mlp<<<(P + TILE_ROWS - 1) / TILE_ROWS, 256, smemBytes, stream>>>(
        ws_feat, ws_wt, ws_bias, ws_seg, ws_venc, (float*)d_out, P);
}
